// PLRNNModule_44727789421025
// MI455X (gfx1250) — matrix operand order repaired and checked by reading, not compiled, not run on hardware
//
#include <hip/hip_runtime.h>
#include <hip/hip_bf16.h>
#include <math.h>

typedef __attribute__((ext_vector_type(16))) __bf16 v16bf;
typedef __attribute__((ext_vector_type(8)))  float  v8f;

#define B_DIM   256
#define T_DIM   2048
#define DZ_DIM  256
#define CLIPV   5.0f
#define ZSTRIDE 260   // 260 mod 64 = 4 -> conflict-reduced LDS column access

// ---------------------------------------------------------------------------
// Prep: stage W = AW with zeroed diagonal into WMMA B-fragment order (bf16).
// Fragment (kt,nt) covers K=kt*32..+31, N=nt*16..+15.
// B-fragment k order used here (contiguous, same as A): lane = half*16 + n_local, element e: k_local =
// half*16 + e.  Flat index = ((kt*16 + nt)*32 + lane)*16 + e
// ---------------------------------------------------------------------------
__global__ void plrnn_prep_w(const float* __restrict__ AW,
                             __bf16* __restrict__ wfrags) {
    int idx  = blockIdx.x * 256 + threadIdx.x;     // 0 .. 65535
    int e    = idx & 15;
    int lane = (idx >> 4) & 31;
    int frag = idx >> 9;                           // kt*16 + nt
    int nt   = frag & 15;
    int kt   = frag >> 4;
    int n    = nt * 16 + (lane & 15);
    int k    = kt * 32 + ((lane >> 4) * 16) + e;
    float v  = (n == k) ? 0.0f : AW[n * DZ_DIM + k];
    wfrags[idx] = (__bf16)v;
}

// ---------------------------------------------------------------------------
// Main: one block = 16 batch rows, 8 waves (256 threads).
// Wave w owns output columns n = w*32 .. w*32+31 (two 16x16 D tiles).
// W fragments are register-resident (128 VGPRs/wave).  z state lives in
// registers in D-fragment layout; LDS carries only relu(z) (f32, for the
// cross-wave transpose into bf16 A-fragments) plus the tiny y/x staging.
// ---------------------------------------------------------------------------
__global__ void __launch_bounds__(256, 1)
plrnn_scan_kernel(const float* __restrict__ xseq,   // (B,T,1)
                  const float* __restrict__ z0,     // (B,DZ)
                  const float* __restrict__ AW,     // (DZ,DZ)
                  const float* __restrict__ h,      // (DZ,)
                  const float* __restrict__ Cw,     // (DZ,1)
                  const float* __restrict__ Rw,     // (1,DZ)
                  const float* __restrict__ Rb,     // (1,)
                  const __bf16* __restrict__ wfrags,
                  float* __restrict__ out) {        // (B,T,1)
    __shared__ float  z_lds[16 * ZSTRIDE];         // relu(z), f32
    __shared__ __bf16 a_lds[8 * 32 * 16];          // bf16 A fragments
    __shared__ float  ysum[16];
    __shared__ float  xcur[16];

    const int tid  = threadIdx.x;
    const int wave = tid >> 5;
    const int lane = tid & 31;
    const int m0   = blockIdx.x * 16;

    const int nA = wave * 32 + (lane & 15);
    const int nB = nA + 16;
    const int mbase = (lane >> 4) * 8;

    const float adA = AW[nA * DZ_DIM + nA];
    const float adB = AW[nB * DZ_DIM + nB];
    const float hA  = h[nA],  hB = h[nB];
    const float cA  = Cw[nA], cB = Cw[nB];
    const float rA  = Rw[nA], rB = Rw[nB];
    const float rb0 = Rb[0];

    // Register-resident W fragments (B operands).
    const v16bf* wsrc = (const v16bf*)wfrags;
    v16bf wfrag[8][2];
#pragma unroll
    for (int kt = 0; kt < 8; ++kt)
#pragma unroll
        for (int j = 0; j < 2; ++j)
            wfrag[kt][j] = wsrc[(kt * 16 + (wave * 2 + j)) * 32 + lane];

    // z state in registers, D-fragment layout: zr0 -> col nA, zr1 -> col nB.
    float zr0[8], zr1[8];
#pragma unroll
    for (int v = 0; v < 8; ++v) {
        zr0[v] = z0[(m0 + mbase + v) * DZ_DIM + nA];
        zr1[v] = z0[(m0 + mbase + v) * DZ_DIM + nB];
    }

    // Seed relu(z0) staging image.
    {
        int m  = tid >> 4;
        int c0 = (tid & 15) * 16;
#pragma unroll
        for (int i = 0; i < 16; ++i)
            z_lds[m * ZSTRIDE + c0 + i] =
                fmaxf(z0[(m0 + m) * DZ_DIM + c0 + i], 0.0f);
    }
    __syncthreads();

    for (int t = 0; t < T_DIM; ++t) {
        // ---- phase 1: re-encode relu(z) into bf16 A-frags (wave w -> kt=w) --
        {
            int m     = lane & 15;
            int kbase = wave * 32 + (lane >> 4) * 16;
            v16bf af;
#pragma unroll
            for (int e = 0; e < 16; ++e) {
                int k = kbase + e;
                af[e] = (__bf16)z_lds[m * ZSTRIDE + k];
            }
            *((v16bf*)(a_lds + (wave * 32 + lane) * 16)) = af;
        }
        if (tid < 16) {
            ysum[tid] = rb0;
            xcur[tid] = xseq[(m0 + tid) * T_DIM + t];
        }
        __syncthreads();

        // ---- phase 2: prefetch all A-frags, then a tight 16-WMMA chain ----
        v16bf a[8];
#pragma unroll
        for (int kt = 0; kt < 8; ++kt)
            a[kt] = *((const v16bf*)(a_lds + (kt * 32 + lane) * 16));

        v8f acc0 = {};
        v8f acc1 = {};
#pragma unroll
        for (int kt = 0; kt < 8; ++kt) {
            acc0 = __builtin_amdgcn_wmma_f32_16x16x32_bf16(
                false, a[kt], false, wfrag[kt][0], (short)0, acc0, false, false);
            acc1 = __builtin_amdgcn_wmma_f32_16x16x32_bf16(
                false, a[kt], false, wfrag[kt][1], (short)0, acc1, false, false);
        }

        // x for this lane's 8 rows: one contiguous 32B LDS read.
        v8f xv = *((const v8f*)(xcur + mbase));

        // ---- f32 state update: z in registers, relu(z) back to LDS ----
#pragma unroll
        for (int v = 0; v < 8; ++v) {
            int   m   = mbase + v;
            float x   = xv[v];
            float zn0 = adA * zr0[v] + acc0[v] + hA + x * cA;
            float zn1 = adB * zr1[v] + acc1[v] + hB + x * cB;
            zn0 = fminf(fmaxf(zn0, -CLIPV), CLIPV);
            zn1 = fminf(fmaxf(zn1, -CLIPV), CLIPV);
            zr0[v] = zn0;
            zr1[v] = zn1;
            z_lds[m * ZSTRIDE + nA] = fmaxf(zn0, 0.0f);
            z_lds[m * ZSTRIDE + nB] = fmaxf(zn1, 0.0f);
            // DY==1 readout: fire-and-forget LDS f32 atomic (no return).
            atomicAdd(&ysum[m], zn0 * rA + zn1 * rB);
        }
        __syncthreads();

        // ---- phase 3: softplus + store (16 lanes) ----
        if (tid < 16) {
            float s = ysum[tid];
            out[(m0 + tid) * T_DIM + t] =
                (s > 20.0f) ? s : log1pf(__expf(s));
        }
    }
}

extern "C" void kernel_launch(void* const* d_in, const int* in_sizes, int n_in,
                              void* d_out, int out_size, void* d_ws, size_t ws_size,
                              hipStream_t stream) {
    const float* xseq = (const float*)d_in[0];   // input_sequence (B,T,1)
    const float* z0   = (const float*)d_in[1];   // (B,DZ)
    const float* AW   = (const float*)d_in[2];   // (DZ,DZ)
    const float* h    = (const float*)d_in[3];   // (DZ,)
    const float* Cw   = (const float*)d_in[4];   // (DZ,1)
    const float* Rw   = (const float*)d_in[5];   // (1,DZ)
    const float* Rb   = (const float*)d_in[6];   // (1,)
    float* out        = (float*)d_out;

    __bf16* wfrags = (__bf16*)d_ws;              // 64K bf16 = 128 KB staged W

    plrnn_prep_w<<<DZ_DIM, 256, 0, stream>>>(AW, wfrags);

    plrnn_scan_kernel<<<B_DIM / 16, 256, 0, stream>>>(
        xseq, z0, AW, h, Cw, Rw, Rb, wfrags, out);
}
